// SplitInjectionAttentionProcessor_21045339750512
// MI455X (gfx1250) — matrix operand order repaired and checked by reading, not compiled, not run on hardware
//
#include <hip/hip_runtime.h>
#include <hip/hip_bf16.h>

typedef _Float16 h16;
typedef __attribute__((ext_vector_type(16))) _Float16 v16h;
typedef __attribute__((ext_vector_type(8)))  _Float16 v8h;
typedef __attribute__((ext_vector_type(8)))  float    v8f;

#define B_  32
#define S_  1024
#define D_  1280
#define CR_ 768
#define H_  20
#define HD_ 64
#define M_  (B_ * S_)          // 32768 rows
#define SCALE_C 0.125f         // 1/sqrt(64)

union V16U { v16h v; v8h h[2]; };

__device__ __forceinline__ v8f wmma_f16(v16h a, v16h b, v8f c) {
  return __builtin_amdgcn_wmma_f32_16x16x32_f16(false, a, false, b, (short)0, c,
                                                false, false);
}

// A fragment 16(M)x32(K), f16.  lanes 0-15 -> M=l, K=0..7 & 16..23;
// lanes 16-31 -> M=l-16, K=8..15 & 24..31.  Two aligned 16B chunks per lane.
__device__ __forceinline__ v16h load_a_frag(const h16* base, int ld) {
  const int l  = threadIdx.x & 31;
  const int m  = l & 15;
  const int kb = (l < 16) ? 0 : 8;
  const h16* p = base + (size_t)m * ld + kb;
  V16U u;
  u.h[0] = *(const v8h*)(p);
  u.h[1] = *(const v8h*)(p + 16);
  return u.v;
}

// B fragment 32(K)x16(N), column n contiguous in K.
// Lanes 0-15 -> N=l, K=0..7 & 16..23; lanes 16-31 -> N=l-16, K=8..15 & 24..31.
__device__ __forceinline__ v16h load_b_frag(const h16* base, int ld) {
  const int l  = threadIdx.x & 31;
  const int n  = l & 15;
  const int kb = (l < 16) ? 0 : 8;
  const h16* p = base + (size_t)n * ld + kb;
  V16U u;
  u.h[0] = *(const v8h*)(p);
  u.h[1] = *(const v8h*)(p + 16);
  return u.v;
}

// xor-butterfly across lanes within groups of 16 (and=0x1f, or=0, xor=mask)
#define SWZ(x, mask) \
  __builtin_bit_cast(float, __builtin_amdgcn_ds_swizzle( \
      __builtin_bit_cast(int, (x)), (0x1F | ((mask) << 10))))

// ---------------- async global->LDS staging (gfx1250 ASYNCcnt path) -------
#if defined(__has_builtin)
#if __has_builtin(__builtin_amdgcn_global_load_async_to_lds_b128)
#define ASYNC_LDS 1
#endif
#endif

typedef int __attribute__((ext_vector_type(4))) i4;
typedef __attribute__((address_space(1))) i4 gi4;   // global int4
typedef __attribute__((address_space(3))) i4 li4;   // LDS int4

__device__ __forceinline__ void g2l_b128(h16* lds, const h16* g) {
#ifdef ASYNC_LDS
  __builtin_amdgcn_global_load_async_to_lds_b128(
      (gi4*)(uintptr_t)g, (li4*)(unsigned)(uintptr_t)lds, 0, 0);
#else
  *(v8h*)lds = *(const v8h*)g;
#endif
}

template <int N>
__device__ __forceinline__ void wait_async() {
#ifdef ASYNC_LDS
#if __has_builtin(__builtin_amdgcn_s_wait_asynccnt)
  __builtin_amdgcn_s_wait_asynccnt(N);
#else
  asm volatile("s_wait_asynccnt %0" ::"i"(N) : "memory");
#endif
#endif
}

// stage 32 contiguous f16 (64B) global->LDS (4 async b128 per thread)
__device__ __forceinline__ void stage_cp32(h16* dst, const h16* src) {
#pragma unroll
  for (int j = 0; j < 4; ++j) g2l_b128(dst + j * 8, src + j * 8);
}

// stage 32 contiguous f32 -> f16 into LDS (conversion must go via VGPRs)
__device__ __forceinline__ void stage_cvt32(h16* dst, const float* src) {
#pragma unroll
  for (int j4 = 0; j4 < 8; ++j4) {
    float4 v = *(const float4*)(src + j4 * 4);
    dst[j4 * 4 + 0] = (h16)v.x;
    dst[j4 * 4 + 1] = (h16)v.y;
    dst[j4 * 4 + 2] = (h16)v.z;
    dst[j4 * 4 + 3] = (h16)v.w;
  }
}

// ---------------------------------------------------------------- utilities
__global__ __launch_bounds__(256) void f32_to_f16_kernel(
    const float* __restrict__ src, h16* __restrict__ dst, int n) {
  int i = blockIdx.x * 256 + threadIdx.x;
  if (i < n) dst[i] = (h16)src[i];
}

__global__ __launch_bounds__(256) void gates_kernel(
    const float* __restrict__ enc, const float* __restrict__ w_gate,
    const float* __restrict__ b_gate, const float* __restrict__ anat_logit,
    const float* __restrict__ dis_logit, float* __restrict__ gates) {
  __shared__ float red[256];
  int b = blockIdx.x, t = threadIdx.x;
  float partial = 0.f;
  for (int c = t; c < CR_; c += 256) {
    float s = 0.f;
    for (int tok = 0; tok < 16; ++tok)           // dis tokens
      s += enc[((size_t)b * 48 + tok) * CR_ + c];
    partial += (s * (1.0f / 16.0f)) * w_gate[c];
  }
  red[t] = partial;
  __syncthreads();
  for (int off = 128; off > 0; off >>= 1) {
    if (t < off) red[t] += red[t + off];
    __syncthreads();
  }
  if (t == 0) {
    float shift = red[0] + b_gate[0];
    gates[b * 2 + 0] = 1.0f / (1.0f + __expf(-(anat_logit[0] - shift)));
    gates[b * 2 + 1] = 1.0f / (1.0f + __expf(-(dis_logit[0] + shift)));
  }
}

// -------------------- KV projection as a WMMA GEMM ------------------------
// M=1024 token rows (0..511 anat -> w_anat, 512..1023 dis -> w_dis),
// N=1280, K=768.  Scatter epilogue into attention layouts:
//   is_v==0: Kcat[b][h][tcat][d]     is_v==1: Vt[b][h][d][tcat]
__global__ __launch_bounds__(256) void kv_gemm_kernel(
    const float* __restrict__ enc, const float* __restrict__ w_anat,
    const float* __restrict__ w_dis, h16* __restrict__ outp, int is_v) {
  __shared__ h16 As[128][80];
  __shared__ h16 Bs[128][80];
  const int tid = threadIdx.x;
  const int n0 = blockIdx.x * 128;   // ho
  const int m0 = blockIdx.y * 128;   // token row
  const int w = tid >> 5, l = tid & 31;
  const int srow = tid >> 1, sseg = (tid & 1) * 32;
  const float* Bsrc = (m0 < 512) ? w_anat : w_dis;

  const int arow = m0 + srow;
  const int ty = arow >> 9, bb = (arow >> 4) & 31, tt = arow & 15;
  // encoder: [:, :16] = dis tokens, [:, 16:32] = anat tokens
  const float* tokrow = enc + ((size_t)bb * 48 + (ty ? tt : 16 + tt)) * CR_;

  v8f acc[8];
#pragma unroll
  for (int nt = 0; nt < 8; ++nt)
#pragma unroll
    for (int e = 0; e < 8; ++e) acc[nt][e] = 0.0f;

  for (int kk = 0; kk < CR_; kk += 64) {
    stage_cvt32(&As[srow][sseg], tokrow + kk + sseg);
    stage_cvt32(&Bs[srow][sseg], Bsrc + (size_t)(n0 + srow) * CR_ + kk + sseg);
    __syncthreads();
#pragma unroll
    for (int kc = 0; kc < 64; kc += 32) {
      v16h af = load_a_frag(&As[w * 16][kc], 80);
      v16h bf[8];
#pragma unroll
      for (int nt = 0; nt < 8; ++nt) bf[nt] = load_b_frag(&Bs[nt * 16][kc], 80);
#pragma unroll
      for (int nt = 0; nt < 8; ++nt) acc[nt] = wmma_f16(af, bf[nt], acc[nt]);
    }
    __syncthreads();
  }
  const int rowoff = (l < 16) ? 0 : 8, cc = l & 15;
#pragma unroll
  for (int nt = 0; nt < 8; ++nt)
#pragma unroll
    for (int r = 0; r < 8; ++r) {
      int m  = m0 + w * 16 + r + rowoff;
      int t2 = m & 15, b2 = (m >> 4) & 31;
      int tcat = (m < 512) ? t2 : 16 + t2;        // 0-15 anat, 16-31 dis
      int ho = n0 + nt * 16 + cc, h = ho >> 6, d = ho & 63;
      h16 val = (h16)acc[nt][r];
      if (!is_v)
        outp[(((size_t)b2 * H_ + h) * 32 + tcat) * 64 + d] = val;
      else
        outp[(((size_t)b2 * H_ + h) * 64 + d) * 32 + tcat] = val;
    }
}

// ------------------------------------------------ GEMM1: Q = X @ Wq^T (f16)
// grid (N/128=10, M/128=256): x = n-tile fastest -> A tile L2-reused 10x.
// Double-buffered pipeline: A prefetched through registers (f32->f16 cvt),
// B staged via async global->LDS.
__global__ __launch_bounds__(256) void gemm_qproj_kernel(
    const float* __restrict__ A, const h16* __restrict__ Bw,
    h16* __restrict__ C) {
  __shared__ h16 As[2][128][80];
  __shared__ h16 Bs[2][128][80];
  const int tid = threadIdx.x;
  const int n0 = blockIdx.x * 128, m0 = blockIdx.y * 128;
  const int w = tid >> 5, l = tid & 31;
  const int srow = tid >> 1, sseg = (tid & 1) * 32;
  const float* arow_ptr = A + (size_t)(m0 + srow) * D_ + sseg;
  const h16*   brow_ptr = Bw + (size_t)(n0 + srow) * D_ + sseg;

  v8f acc[8];
#pragma unroll
  for (int nt = 0; nt < 8; ++nt)
#pragma unroll
    for (int e = 0; e < 8; ++e) acc[nt][e] = 0.0f;

  float4 areg[8];
#pragma unroll
  for (int j4 = 0; j4 < 8; ++j4)
    areg[j4] = *(const float4*)(arow_ptr + j4 * 4);
  stage_cp32(&Bs[0][srow][sseg], brow_ptr);

  const int NIT = D_ / 64;  // 20
  for (int it = 0; it < NIT; ++it) {
    const int cur = it & 1;
    h16* ad = &As[cur][srow][sseg];
#pragma unroll
    for (int j4 = 0; j4 < 8; ++j4) {
      ad[j4 * 4 + 0] = (h16)areg[j4].x;
      ad[j4 * 4 + 1] = (h16)areg[j4].y;
      ad[j4 * 4 + 2] = (h16)areg[j4].z;
      ad[j4 * 4 + 3] = (h16)areg[j4].w;
    }
    if (it + 1 < NIT) {
      const int kk = (it + 1) * 64;
#pragma unroll
      for (int j4 = 0; j4 < 8; ++j4)
        areg[j4] = *(const float4*)(arow_ptr + kk + j4 * 4);
      stage_cp32(&Bs[cur ^ 1][srow][sseg], brow_ptr + kk);
      wait_async<4>();   // in-order: current tile's 4 asyncs complete
    } else {
      wait_async<0>();
    }
    __syncthreads();
#pragma unroll
    for (int kc = 0; kc < 64; kc += 32) {
      v16h af = load_a_frag(&As[cur][w * 16][kc], 80);
      v16h bf[8];
#pragma unroll
      for (int nt = 0; nt < 8; ++nt)
        bf[nt] = load_b_frag(&Bs[cur][nt * 16][kc], 80);
#pragma unroll
      for (int nt = 0; nt < 8; ++nt) acc[nt] = wmma_f16(af, bf[nt], acc[nt]);
    }
    __syncthreads();
  }
  const int rowoff = (l < 16) ? 0 : 8, cc = l & 15;
#pragma unroll
  for (int nt = 0; nt < 8; ++nt)
#pragma unroll
    for (int r = 0; r < 8; ++r)
      C[(size_t)(m0 + w * 16 + r + rowoff) * D_ + (n0 + nt * 16 + cc)] =
          (h16)acc[nt][r];
}

// ------------------------------- fused attention (scores+softmax+gate+PV)
__global__ __launch_bounds__(256) void attention_kernel(
    const h16* __restrict__ Q, const h16* __restrict__ Kcat,
    const h16* __restrict__ Vt, const float* __restrict__ gates,
    h16* __restrict__ Z) {
  __shared__ h16 Plds[8][16][32];
  const int tid = threadIdx.x, w = tid >> 5, l = tid & 31;
  const int bid = blockIdx.x;
  const int grp = bid & 7;
  const int h   = (bid >> 3) % H_;
  const int b   = bid / (8 * H_);
  const int s0  = (grp * 8 + w) * 16;

  const h16* qbase = Q + ((size_t)(b * S_ + s0)) * D_ + h * HD_;
  v16h q0 = load_a_frag(qbase, D_);        // K = 0..31
  v16h q1 = load_a_frag(qbase + 32, D_);   // K = 32..63

  const h16* kbase = Kcat + (size_t)(b * H_ + h) * 32 * 64;
  v16h kb00 = load_b_frag(kbase + 0 * 16 * 64 + 0, 64);   // anat, d 0..31
  v16h kb01 = load_b_frag(kbase + 0 * 16 * 64 + 32, 64);  // anat, d 32..63
  v16h kb10 = load_b_frag(kbase + 16 * 64 + 0, 64);       // dis,  d 0..31
  v16h kb11 = load_b_frag(kbase + 16 * 64 + 32, 64);      // dis,  d 32..63

  v8f sa, sd;
#pragma unroll
  for (int e = 0; e < 8; ++e) { sa[e] = 0.f; sd[e] = 0.f; }
  sa = wmma_f16(q0, kb00, sa);
  sa = wmma_f16(q1, kb01, sa);
  sd = wmma_f16(q0, kb10, sd);
  sd = wmma_f16(q1, kb11, sd);

  const float ga = gates[b * 2 + 0], gd = gates[b * 2 + 1];
  const int rowoff = (l < 16) ? 0 : 8, cc = l & 15;

  // softmax over 16 tokens: columns live across a group of 16 lanes
#pragma unroll
  for (int r = 0; r < 8; ++r) {
    float x = sa[r] * SCALE_C;
    float mx = x;
    mx = fmaxf(mx, SWZ(mx, 1)); mx = fmaxf(mx, SWZ(mx, 2));
    mx = fmaxf(mx, SWZ(mx, 4)); mx = fmaxf(mx, SWZ(mx, 8));
    float e = __expf(x - mx);
    float s = e;
    s += SWZ(s, 1); s += SWZ(s, 2); s += SWZ(s, 4); s += SWZ(s, 8);
    Plds[w][r + rowoff][cc] = (h16)((e / s) * ga);

    float xd = sd[r] * SCALE_C;
    float md = xd;
    md = fmaxf(md, SWZ(md, 1)); md = fmaxf(md, SWZ(md, 2));
    md = fmaxf(md, SWZ(md, 4)); md = fmaxf(md, SWZ(md, 8));
    float ed = __expf(xd - md);
    float sdm = ed;
    sdm += SWZ(sdm, 1); sdm += SWZ(sdm, 2); sdm += SWZ(sdm, 4); sdm += SWZ(sdm, 8);
    Plds[w][r + rowoff][16 + cc] = (h16)((ed / sdm) * gd);
  }
  // per-wave LDS region; DS ops are in-order within a wave -> no barrier

  v16h pf = load_a_frag(&Plds[w][0][0], 32);   // 16 x 32 gated probabilities
  const h16* vbase = Vt + (size_t)(b * H_ + h) * 64 * 32;
  h16* zout = Z + ((size_t)(b * S_ + s0)) * D_ + h * HD_;
#pragma unroll
  for (int nt = 0; nt < 4; ++nt) {
    v16h vb = load_b_frag(vbase + nt * 16 * 32, 32);
    v8f z;
#pragma unroll
    for (int e = 0; e < 8; ++e) z[e] = 0.f;
    z = wmma_f16(pf, vb, z);   // ga*Pa@Va + gd*Pd@Vd in one WMMA (K=32)
#pragma unroll
    for (int r = 0; r < 8; ++r)
      zout[(size_t)(r + rowoff) * D_ + nt * 16 + cc] = (h16)z[r];
  }
}

// ------------------- GEMM2: out = Z @ Wout^T + b_out + residual (f32 out)
// Double-buffered: both operands staged via async global->LDS.
__global__ __launch_bounds__(256) void gemm_out_kernel(
    const h16* __restrict__ A, const h16* __restrict__ Bw,
    const float* __restrict__ bias, const float* __restrict__ resid,
    float* __restrict__ out) {
  __shared__ h16 As[2][128][80];
  __shared__ h16 Bs[2][128][80];
  const int tid = threadIdx.x;
  const int n0 = blockIdx.x * 128, m0 = blockIdx.y * 128;
  const int w = tid >> 5, l = tid & 31;
  const int srow = tid >> 1, sseg = (tid & 1) * 32;
  const h16* arow_ptr = A + (size_t)(m0 + srow) * D_ + sseg;
  const h16* brow_ptr = Bw + (size_t)(n0 + srow) * D_ + sseg;

  v8f acc[8];
#pragma unroll
  for (int nt = 0; nt < 8; ++nt)
#pragma unroll
    for (int e = 0; e < 8; ++e) acc[nt][e] = 0.0f;

  stage_cp32(&As[0][srow][sseg], arow_ptr);
  stage_cp32(&Bs[0][srow][sseg], brow_ptr);

  const int NIT = D_ / 64;  // 20
  for (int it = 0; it < NIT; ++it) {
    const int cur = it & 1;
    if (it + 1 < NIT) {
      const int kk = (it + 1) * 64;
      stage_cp32(&As[cur ^ 1][srow][sseg], arow_ptr + kk);
      stage_cp32(&Bs[cur ^ 1][srow][sseg], brow_ptr + kk);
      wait_async<8>();   // in-order: current tile's 8 asyncs complete
    } else {
      wait_async<0>();
    }
    __syncthreads();
#pragma unroll
    for (int kc = 0; kc < 64; kc += 32) {
      v16h af = load_a_frag(&As[cur][w * 16][kc], 80);
      v16h bf[8];
#pragma unroll
      for (int nt = 0; nt < 8; ++nt)
        bf[nt] = load_b_frag(&Bs[cur][nt * 16][kc], 80);
#pragma unroll
      for (int nt = 0; nt < 8; ++nt) acc[nt] = wmma_f16(af, bf[nt], acc[nt]);
    }
    __syncthreads();
  }
  const int rowoff = (l < 16) ? 0 : 8, cc = l & 15;
#pragma unroll
  for (int nt = 0; nt < 8; ++nt)
#pragma unroll
    for (int r = 0; r < 8; ++r) {
      size_t row = (size_t)(m0 + w * 16 + r + rowoff);
      int    col = n0 + nt * 16 + cc;
      out[row * D_ + col] = acc[nt][r] + bias[col] + resid[row * D_ + col];
    }
}

// ---------------------------------------------------------------- launcher
extern "C" void kernel_launch(void* const* d_in, const int* in_sizes, int n_in,
                              void* d_out, int out_size, void* d_ws,
                              size_t ws_size, hipStream_t stream) {
  const float* hidden  = (const float*)d_in[0];
  const float* enc     = (const float*)d_in[1];
  const float* w_q     = (const float*)d_in[2];
  const float* w_k     = (const float*)d_in[3];
  const float* w_v     = (const float*)d_in[4];
  const float* w_k_dis = (const float*)d_in[5];
  const float* w_v_dis = (const float*)d_in[6];
  const float* w_gate  = (const float*)d_in[7];
  const float* b_gate  = (const float*)d_in[8];
  const float* anat_lg = (const float*)d_in[9];
  const float* dis_lg  = (const float*)d_in[10];
  const float* w_out   = (const float*)d_in[11];
  const float* b_out   = (const float*)d_in[12];
  float* out = (float*)d_out;

  char* ws = (char*)d_ws;
  size_t off = 0;
  auto take = [&](size_t bytes) {
    char* p = ws + off;
    off = (off + bytes + 255) & ~(size_t)255;
    return p;
  };
  h16*   wq16   = (h16*)take((size_t)D_ * D_ * 2);
  h16*   wout16 = (h16*)take((size_t)D_ * D_ * 2);
  h16*   Kcat   = (h16*)take((size_t)B_ * H_ * 32 * 64 * 2);
  h16*   Vt     = (h16*)take((size_t)B_ * H_ * 64 * 32 * 2);
  float* gates  = (float*)take((size_t)B_ * 2 * 4);
  h16*   Q16    = (h16*)take((size_t)M_ * D_ * 2);
  h16*   Z16    = (h16*)take((size_t)M_ * D_ * 2);
  (void)ws_size; (void)in_sizes; (void)n_in; (void)out_size;

  const int nw = D_ * D_;  // 1,638,400
  f32_to_f16_kernel<<<(nw + 255) / 256, 256, 0, stream>>>(w_q, wq16, nw);
  f32_to_f16_kernel<<<(nw + 255) / 256, 256, 0, stream>>>(w_out, wout16, nw);
  kv_gemm_kernel<<<dim3(D_ / 128, 8), 256, 0, stream>>>(enc, w_k, w_k_dis,
                                                        Kcat, 0);
  kv_gemm_kernel<<<dim3(D_ / 128, 8), 256, 0, stream>>>(enc, w_v, w_v_dis,
                                                        Vt, 1);
  gates_kernel<<<B_, 256, 0, stream>>>(enc, w_gate, b_gate, anat_lg, dis_lg,
                                       gates);
  gemm_qproj_kernel<<<dim3(D_ / 128, M_ / 128), 256, 0, stream>>>(hidden, wq16,
                                                                  Q16);
  attention_kernel<<<B_ * H_ * 8, 256, 0, stream>>>(Q16, Kcat, Vt, gates, Z16);
  gemm_out_kernel<<<dim3(D_ / 128, M_ / 128), 256, 0, stream>>>(
      Z16, wout16, b_out, hidden, out);
}
